// Attention_GlobalCLS_63436666961993
// MI455X (gfx1250) — compile-verified, matrix operand order checked by reading, not run on hardware
//
#include <hip/hip_runtime.h>
#include <hip/hip_bf16.h>
#include <math.h>

typedef __attribute__((ext_vector_type(16))) _Float16 v16h;
typedef __attribute__((ext_vector_type(8)))  float    v8f;

#define EMBED 768
#define NH    12
#define HD    64
#define QKVD  2304
#define TGT   197
#define BSZ   128
#define NFR   16
#define BG    8          // BSZ / NFR
#define SEQG  (NFR*TGT)  // 3152
#define SCALE 0.125f     // 64^-0.5
#define NEGINF (-1.0e30f)

// ---------------------------------------------------------------------------
// WMMA helpers (CDNA5 wave32, V_WMMA_F32_16X16X32_F16)
// ---------------------------------------------------------------------------
__device__ __forceinline__ v8f wmma32(v16h a, v16h b, v8f c) {
  return __builtin_amdgcn_wmma_f32_16x16x32_f16(false, a, false, b, (short)0, c,
                                                false, false);
}

// 16-bit A-matrix 16x32 per-lane fragment: lane holds row (lane&15); lanes
// 0-15 carry K = [0..7] and [16..23], lanes 16-31 carry K = [8..15],[24..31].
__device__ __forceinline__ v16h frag_rowptr(const _Float16* rowp, int kb8) {
  v16h f;
#pragma unroll
  for (int i = 0; i < 8; ++i) {
    f[i]     = rowp[kb8 + i];
    f[i + 8] = rowp[kb8 + 16 + i];
  }
  return f;
}

__device__ __forceinline__ v16h frag_rm(const _Float16* base, int ld, int l15,
                                        int kb8) {
  return frag_rowptr(base + (size_t)l15 * ld, kb8);
}

// Cross-lane xor reduction via DS_SWIZZLE (group-of-32: xor_mask<<10 | and=31)
#define SWZ_XOR(x, m) \
  __int_as_float(__builtin_amdgcn_ds_swizzle(__float_as_int(x), ((m) << 10) | 31))
__device__ __forceinline__ float red_max16(float v) {
  v = fmaxf(v, SWZ_XOR(v, 1));
  v = fmaxf(v, SWZ_XOR(v, 2));
  v = fmaxf(v, SWZ_XOR(v, 4));
  v = fmaxf(v, SWZ_XOR(v, 8));
  return v;
}
__device__ __forceinline__ float red_sum16(float v) {
  v += SWZ_XOR(v, 1);
  v += SWZ_XOR(v, 2);
  v += SWZ_XOR(v, 4);
  v += SWZ_XOR(v, 8);
  return v;
}

// Async global->LDS copy (GLOBAL_LOAD_ASYNC_TO_LDS_B128, ASYNCcnt tracked).
// ldsoff: wave-relative LDS byte offset (low 32 bits of a shared pointer).
__device__ __forceinline__ void async_g2l_b128(unsigned ldsoff,
                                               const void* gptr) {
  asm volatile("global_load_async_to_lds_b128 %0, %1, off"
               :: "v"(ldsoff), "v"(gptr)
               : "memory");
}
__device__ __forceinline__ void wait_async0() {
  asm volatile("s_wait_asynccnt 0" ::: "memory");
}

// ---------------------------------------------------------------------------
// Small prep kernels
// ---------------------------------------------------------------------------
__global__ void conv_f32_f16(const float* __restrict__ s,
                             _Float16* __restrict__ d, int n) {
  int i = blockIdx.x * 256 + threadIdx.x;
  if (i < n) d[i] = (_Float16)s[i];
}

// W_eff[i][j] = in_proj_weight[i][j] + sum_r lora_b[i][r] * lora_a[r][j]
__global__ void weff_kernel(const float* __restrict__ w,
                            const float* __restrict__ la,
                            const float* __restrict__ lb,
                            _Float16* __restrict__ out) {
  int idx = blockIdx.x * 256 + threadIdx.x;  // QKVD*EMBED exact
  int i = idx / EMBED, j = idx - i * EMBED;
  float acc = w[idx];
#pragma unroll 8
  for (int r = 0; r < 64; ++r) acc += lb[i * 64 + r] * la[r * EMBED + j];
  out[idx] = (_Float16)acc;
}

__global__ void gather_q1(const _Float16* __restrict__ qkv,
                          const int* __restrict__ top_idx,
                          _Float16* __restrict__ q1) {
  int idx = blockIdx.x * 256 + threadIdx.x;  // BSZ*EMBED exact
  int bf = idx / EMBED, c = idx - bf * EMBED;
  int t = top_idx[bf];
  q1[idx] = qkv[((size_t)bf * TGT + t) * QKVD + c];  // q section: cols [0,768)
}

// ---------------------------------------------------------------------------
// Tiled WMMA GEMM: C[M,N] = A[M,K] @ B[N,K]^T + bias[N]
// Block = 256 threads (8 waves), 64x64 tile; K slabs of 32 staged into LDS
// via async global->LDS (no VGPR round trip). M,N mult of 64; K mult of 32.
// ---------------------------------------------------------------------------
template <bool F32OUT>
__global__ __launch_bounds__(256) void gemm_absT(
    const _Float16* __restrict__ A, const _Float16* __restrict__ B,
    const float* __restrict__ bias, void* __restrict__ Cout, int M, int N,
    int K) {
  __shared__ _Float16 As[64 * 32];
  __shared__ _Float16 Bs[64 * 32];
  const int m0 = blockIdx.x * 64, n0 = blockIdx.y * 64;
  const int t = threadIdx.x, w = t >> 5, lane = t & 31;
  const int l15 = lane & 15, kb8 = (lane >> 4) << 3;
  const int wr = w & 3, wc = w >> 2;  // wave tile: rows wr*16, cols wc*32
  const int srow = t >> 2, schunk = (t & 3) << 3;
  const unsigned ldsA = (unsigned)(size_t)&As[srow * 32 + schunk];
  const unsigned ldsB = (unsigned)(size_t)&Bs[srow * 32 + schunk];
  v8f acc0 = {}, acc1 = {};
  for (int k0 = 0; k0 < K; k0 += 32) {
    __syncthreads();
    async_g2l_b128(ldsA, &A[(size_t)(m0 + srow) * K + k0 + schunk]);
    async_g2l_b128(ldsB, &B[(size_t)(n0 + srow) * K + k0 + schunk]);
    wait_async0();
    __syncthreads();
    v16h a  = frag_rm(&As[(wr * 16) * 32], 32, l15, kb8);
    v16h b0 = frag_rm(&Bs[(wc * 32) * 32], 32, l15, kb8);
    v16h b1 = frag_rm(&Bs[(wc * 32 + 16) * 32], 32, l15, kb8);
    acc0 = wmma32(a, b0, acc0);
    acc1 = wmma32(a, b1, acc1);
  }
  const int row0 = m0 + wr * 16 + kb8;  // C/D layout: VGPR r -> row kb8 + r
  const int c0 = n0 + wc * 32 + l15, c1 = c0 + 16;
  const float bi0 = bias[c0], bi1 = bias[c1];
#pragma unroll
  for (int r = 0; r < 8; ++r) {
    float v0 = acc0[r] + bi0, v1 = acc1[r] + bi1;
    size_t o0 = (size_t)(row0 + r) * N + c0;
    size_t o1 = (size_t)(row0 + r) * N + c1;
    if (F32OUT) {
      ((float*)Cout)[o0] = v0;
      ((float*)Cout)[o1] = v1;
    } else {
      ((_Float16*)Cout)[o0] = (_Float16)v0;
      ((_Float16*)Cout)[o1] = (_Float16)v1;
    }
  }
}

// ---------------------------------------------------------------------------
// Flash attention, one wave32 per (q-tile of 16, head, batch). seq = 197,
// keys processed 32 at a time (7 chunks, tail masked). Q pre-scaled by
// 1/sqrt(64) so the inner loop only adds the mask. qkv layout: row = b*197+t,
// cols [0,768)=q [768,1536)=k [1536,2304)=v, f16.
// ---------------------------------------------------------------------------
__global__ __launch_bounds__(32) void attn_kernel(
    const _Float16* __restrict__ qkv, const float* __restrict__ mask,
    _Float16* __restrict__ xo) {
  const int qt = blockIdx.x, h = blockIdx.y, bi = blockIdx.z;
  const int lane = threadIdx.x & 31;
  const int l15 = lane & 15, kb8 = (lane >> 4) << 3;
  __shared__ _Float16 P[16 * 32];
  const _Float16* qkvB = qkv + (size_t)bi * TGT * QKVD;

  int qr = qt * 16 + l15;
  int qrc = qr > TGT - 1 ? TGT - 1 : qr;
  const _Float16* qp = qkvB + (size_t)qrc * QKVD + h * HD;
  v16h qa0 = frag_rowptr(qp, kb8);
  v16h qa1 = frag_rowptr(qp + 32, kb8);
#pragma unroll
  for (int i = 0; i < 16; ++i) {  // fold softmax scaling into Q
    qa0[i] = qa0[i] * (_Float16)SCALE;
    qa1[i] = qa1[i] * (_Float16)SCALE;
  }

  float mrow[8], lrow[8];
  v8f o0 = {}, o1 = {}, o2 = {}, o3 = {};
#pragma unroll
  for (int r = 0; r < 8; ++r) { mrow[r] = NEGINF; lrow[r] = 0.f; }

  for (int c = 0; c < 7; ++c) {
    const int kb = c * 32;
    const int key0 = kb + l15, key1 = key0 + 16;
    const int k0c = key0 > TGT - 1 ? TGT - 1 : key0;
    const int k1c = key1 > TGT - 1 ? TGT - 1 : key1;
    const _Float16* kp0 = qkvB + (size_t)k0c * QKVD + EMBED + h * HD;
    const _Float16* kp1 = qkvB + (size_t)k1c * QKVD + EMBED + h * HD;
    v8f z = {};
    v8f s0 = wmma32(qa0, frag_rowptr(kp0, kb8), z);
    s0 = wmma32(qa1, frag_rowptr(kp0 + 32, kb8), s0);
    v8f s1 = wmma32(qa0, frag_rowptr(kp1, kb8), z);
    s1 = wmma32(qa1, frag_rowptr(kp1 + 32, kb8), s1);

    float t0[8], t1[8];
#pragma unroll
    for (int r = 0; r < 8; ++r) {
      int qq = qt * 16 + kb8 + r;
      int qc = qq > TGT - 1 ? TGT - 1 : qq;
      const float* mk = mask + ((size_t)bi * TGT + qc) * TGT;
      t0[r] = (key0 < TGT) ? s0[r] + mk[key0] : NEGINF;
      t1[r] = (key1 < TGT) ? s1[r] + mk[key1] : NEGINF;
    }
    float rs[8], alf[8];
#pragma unroll
    for (int r = 0; r < 8; ++r) {
      float mx = red_max16(fmaxf(t0[r], t1[r]));
      float mn = fmaxf(mrow[r], mx);
      alf[r] = __expf(mrow[r] - mn);
      mrow[r] = mn;
      float p0 = __expf(t0[r] - mn), p1 = __expf(t1[r] - mn);
      rs[r] = p0 + p1;
      P[(kb8 + r) * 32 + l15] = (_Float16)p0;
      P[(kb8 + r) * 32 + 16 + l15] = (_Float16)p1;
    }
#pragma unroll
    for (int r = 0; r < 8; ++r) {
      lrow[r] = lrow[r] * alf[r] + red_sum16(rs[r]);
      o0[r] *= alf[r]; o1[r] *= alf[r]; o2[r] *= alf[r]; o3[r] *= alf[r];
    }
    __syncthreads();
    v16h pa = frag_rm(P, 32, l15, kb8);
#pragma unroll
    for (int nt = 0; nt < 4; ++nt) {
      v16h vb;
#pragma unroll
      for (int i = 0; i < 8; ++i) {
        int ka = kb + kb8 + i;      ka = ka > TGT - 1 ? TGT - 1 : ka;
        int kc = kb + kb8 + 16 + i; kc = kc > TGT - 1 ? TGT - 1 : kc;
        vb[i]     = qkvB[(size_t)ka * QKVD + 2 * EMBED + h * HD + nt * 16 + l15];
        vb[i + 8] = qkvB[(size_t)kc * QKVD + 2 * EMBED + h * HD + nt * 16 + l15];
      }
      if (nt == 0) o0 = wmma32(pa, vb, o0);
      if (nt == 1) o1 = wmma32(pa, vb, o1);
      if (nt == 2) o2 = wmma32(pa, vb, o2);
      if (nt == 3) o3 = wmma32(pa, vb, o3);
    }
    __syncthreads();
  }
#pragma unroll
  for (int r = 0; r < 8; ++r) {
    int row = qt * 16 + kb8 + r;
    if (row < TGT) {
      float inv = 1.f / lrow[r];
      size_t base = ((size_t)bi * TGT + row) * EMBED + h * HD + l15;
      xo[base +  0] = (_Float16)(o0[r] * inv);
      xo[base + 16] = (_Float16)(o1[r] * inv);
      xo[base + 32] = (_Float16)(o2[r] * inv);
      xo[base + 48] = (_Float16)(o3[r] * inv);
    }
  }
}

// ---------------------------------------------------------------------------
// Global-CLS flash attention: one wave per (head, group b). 16 queries, 3152
// contiguous key rows [b*3152,(b+1)*3152). mask_g[j]=mask[b*16+j/197,196,j%197]
// ---------------------------------------------------------------------------
__global__ __launch_bounds__(32) void cls_attn_kernel(
    const _Float16* __restrict__ qkv, const _Float16* __restrict__ q1,
    const float* __restrict__ mask, _Float16* __restrict__ x1) {
  const int h = blockIdx.x, b = blockIdx.y;
  const int lane = threadIdx.x & 31;
  const int l15 = lane & 15, kb8 = (lane >> 4) << 3;
  __shared__ _Float16 P[16 * 32];
  const _Float16* qkvB = qkv + (size_t)b * SEQG * QKVD;
  const _Float16* qp = q1 + ((size_t)(b * NFR) + l15) * EMBED + h * HD;
  v16h qa0 = frag_rowptr(qp, kb8);
  v16h qa1 = frag_rowptr(qp + 32, kb8);
#pragma unroll
  for (int i = 0; i < 16; ++i) {
    qa0[i] = qa0[i] * (_Float16)SCALE;
    qa1[i] = qa1[i] * (_Float16)SCALE;
  }

  float mrow[8], lrow[8];
  v8f o0 = {}, o1 = {}, o2 = {}, o3 = {};
#pragma unroll
  for (int r = 0; r < 8; ++r) { mrow[r] = NEGINF; lrow[r] = 0.f; }

  for (int c = 0; c < 99; ++c) {
    const int kb = c * 32;
    const int key0 = kb + l15, key1 = key0 + 16;
    const int k0c = key0 > SEQG - 1 ? SEQG - 1 : key0;
    const int k1c = key1 > SEQG - 1 ? SEQG - 1 : key1;
    const _Float16* kp0 = qkvB + (size_t)k0c * QKVD + EMBED + h * HD;
    const _Float16* kp1 = qkvB + (size_t)k1c * QKVD + EMBED + h * HD;
    v8f z = {};
    v8f s0 = wmma32(qa0, frag_rowptr(kp0, kb8), z);
    s0 = wmma32(qa1, frag_rowptr(kp0 + 32, kb8), s0);
    v8f s1 = wmma32(qa0, frag_rowptr(kp1, kb8), z);
    s1 = wmma32(qa1, frag_rowptr(kp1 + 32, kb8), s1);

    float mg0 = 0.f, mg1 = 0.f;
    if (key0 < SEQG) {
      int f = key0 / TGT, tt = key0 - f * TGT;
      mg0 = mask[((size_t)(b * NFR + f) * TGT + (TGT - 1)) * TGT + tt];
    }
    if (key1 < SEQG) {
      int f = key1 / TGT, tt = key1 - f * TGT;
      mg1 = mask[((size_t)(b * NFR + f) * TGT + (TGT - 1)) * TGT + tt];
    }
    float t0[8], t1[8];
#pragma unroll
    for (int r = 0; r < 8; ++r) {
      t0[r] = (key0 < SEQG) ? s0[r] + mg0 : NEGINF;
      t1[r] = (key1 < SEQG) ? s1[r] + mg1 : NEGINF;
    }
    float rs[8], alf[8];
#pragma unroll
    for (int r = 0; r < 8; ++r) {
      float mx = red_max16(fmaxf(t0[r], t1[r]));
      float mn = fmaxf(mrow[r], mx);
      alf[r] = __expf(mrow[r] - mn);
      mrow[r] = mn;
      float p0 = __expf(t0[r] - mn), p1 = __expf(t1[r] - mn);
      rs[r] = p0 + p1;
      P[(kb8 + r) * 32 + l15] = (_Float16)p0;
      P[(kb8 + r) * 32 + 16 + l15] = (_Float16)p1;
    }
#pragma unroll
    for (int r = 0; r < 8; ++r) {
      lrow[r] = lrow[r] * alf[r] + red_sum16(rs[r]);
      o0[r] *= alf[r]; o1[r] *= alf[r]; o2[r] *= alf[r]; o3[r] *= alf[r];
    }
    __syncthreads();
    v16h pa = frag_rm(P, 32, l15, kb8);
#pragma unroll
    for (int nt = 0; nt < 4; ++nt) {
      v16h vb;
#pragma unroll
      for (int i = 0; i < 8; ++i) {
        int ka = kb + kb8 + i;      ka = ka > SEQG - 1 ? SEQG - 1 : ka;
        int kc = kb + kb8 + 16 + i; kc = kc > SEQG - 1 ? SEQG - 1 : kc;
        vb[i]     = qkvB[(size_t)ka * QKVD + 2 * EMBED + h * HD + nt * 16 + l15];
        vb[i + 8] = qkvB[(size_t)kc * QKVD + 2 * EMBED + h * HD + nt * 16 + l15];
      }
      if (nt == 0) o0 = wmma32(pa, vb, o0);
      if (nt == 1) o1 = wmma32(pa, vb, o1);
      if (nt == 2) o2 = wmma32(pa, vb, o2);
      if (nt == 3) o3 = wmma32(pa, vb, o3);
    }
    __syncthreads();
  }
#pragma unroll
  for (int r = 0; r < 8; ++r) {
    int f = kb8 + r;  // frame index, all 16 valid
    float inv = 1.f / lrow[r];
    size_t base = ((size_t)(b * NFR) + f) * EMBED + h * HD + l15;
    x1[base +  0] = (_Float16)(o0[r] * inv);
    x1[base + 16] = (_Float16)(o1[r] * inv);
    x1[base + 32] = (_Float16)(o2[r] * inv);
    x1[base + 48] = (_Float16)(o3[r] * inv);
  }
}

// ---------------------------------------------------------------------------
// Per-CLS-row MLP (768->64 quickGELU ->768) + add x1_proj, scatter into out.
// ---------------------------------------------------------------------------
__global__ __launch_bounds__(256) void cls_update(
    float* __restrict__ out, const float* __restrict__ x1p,
    const int* __restrict__ top_idx, const float* __restrict__ dw,
    const float* __restrict__ db, const float* __restrict__ uw,
    const float* __restrict__ ub) {
  __shared__ float xb[EMBED];
  __shared__ float hb[64];
  const int bf = blockIdx.x, t = threadIdx.x;
  const int tok = top_idx[bf];
  float* row = out + ((size_t)bf * TGT + tok) * EMBED;
  for (int c = t; c < EMBED; c += 256) xb[c] = row[c];
  __syncthreads();
  if (t < 64) {
    float a = db[t];
#pragma unroll 8
    for (int j = 0; j < EMBED; ++j) a += dw[t * EMBED + j] * xb[j];
    hb[t] = a / (1.f + __expf(-1.702f * a));  // quick_gelu
  }
  __syncthreads();
  for (int c = t; c < EMBED; c += 256) {
    float a = ub[c];
#pragma unroll
    for (int r = 0; r < 64; ++r) a += uw[c * 64 + r] * hb[r];
    row[c] = a + x1p[(size_t)bf * EMBED + c];
  }
}

// ---------------------------------------------------------------------------
extern "C" void kernel_launch(void* const* d_in, const int* in_sizes, int n_in,
                              void* d_out, int out_size, void* d_ws,
                              size_t ws_size, hipStream_t stream) {
  const float* x      = (const float*)d_in[0];
  const float* amask  = (const float*)d_in[1];
  const float* in_w   = (const float*)d_in[2];
  const float* in_b   = (const float*)d_in[3];
  const float* out_w  = (const float*)d_in[4];
  const float* out_b  = (const float*)d_in[5];
  const float* lora_a = (const float*)d_in[6];
  const float* lora_b = (const float*)d_in[7];
  const float* down_w = (const float*)d_in[8];
  const float* down_b = (const float*)d_in[9];
  const float* up_w   = (const float*)d_in[10];
  const float* up_b   = (const float*)d_in[11];
  const int*   topidx = (const int*)d_in[13];
  float* out = (float*)d_out;

  char* ws = (char*)d_ws;
  size_t off = 0;
  auto carve = [&](size_t bytes) -> void* {
    void* p = ws + off;
    off += (bytes + 255) & ~(size_t)255;
    return p;
  };
  const int M = BSZ * TGT;  // 25216, multiple of 64
  _Float16* Xf   = (_Float16*)carve((size_t)M * EMBED * 2);
  _Float16* Weff = (_Float16*)carve((size_t)QKVD * EMBED * 2);
  _Float16* Wout = (_Float16*)carve((size_t)EMBED * EMBED * 2);
  _Float16* QKV  = (_Float16*)carve((size_t)M * QKVD * 2);
  _Float16* XO   = (_Float16*)carve((size_t)M * EMBED * 2);
  _Float16* Q1   = (_Float16*)carve((size_t)BSZ * EMBED * 2);
  _Float16* X1   = (_Float16*)carve((size_t)BSZ * EMBED * 2);
  float*    X1P  = (float*)carve((size_t)BSZ * EMBED * 4);

  // 1) prep: fold LoRA into weights, f16 conversions
  conv_f32_f16<<<(M * EMBED) / 256, 256, 0, stream>>>(x, Xf, M * EMBED);
  weff_kernel<<<(QKVD * EMBED) / 256, 256, 0, stream>>>(in_w, lora_a, lora_b,
                                                        Weff);
  conv_f32_f16<<<(EMBED * EMBED) / 256, 256, 0, stream>>>(out_w, Wout,
                                                          EMBED * EMBED);
  // 2) qkv = x @ Weff^T + in_proj_bias  (f16 out)
  gemm_absT<false><<<dim3(M / 64, QKVD / 64), 256, 0, stream>>>(
      Xf, Weff, in_b, QKV, M, QKVD, EMBED);
  // 3) per-token attention -> XO (f16)
  attn_kernel<<<dim3(13, NH, BSZ), 32, 0, stream>>>(QKV, amask, XO);
  // 4) out = XO @ Wout^T + out_bias (f32, full output)
  gemm_absT<true><<<dim3(M / 64, EMBED / 64), 256, 0, stream>>>(
      XO, Wout, out_b, out, M, EMBED, EMBED);
  // 5) global CLS path
  gather_q1<<<(BSZ * EMBED) / 256, 256, 0, stream>>>(QKV, topidx, Q1);
  cls_attn_kernel<<<dim3(NH, BG), 32, 0, stream>>>(QKV, Q1, amask, X1);
  gemm_absT<true><<<dim3(BSZ / 64, EMBED / 64), 256, 0, stream>>>(
      X1, Wout, out_b, X1P, BSZ, EMBED, EMBED);
  // 6) MLP on gathered rows + add x1 projection, scatter into out
  cls_update<<<BSZ, 256, 0, stream>>>(out, X1P, topidx, down_w, down_b, up_w,
                                      up_b);
}
